// modelA1_35888746725740
// MI455X (gfx1250) — matrix operand order repaired and checked by reading, not compiled, not run on hardware
//
#include <hip/hip_runtime.h>
#include <hip/hip_bf16.h>

typedef __attribute__((ext_vector_type(16))) _Float16 v16h;
typedef __attribute__((ext_vector_type(8)))  _Float16 h8;
typedef __attribute__((ext_vector_type(2)))  _Float16 h2;
typedef __attribute__((ext_vector_type(8)))  float    v8f;

#define N_NODES 30000
#define N_EDGES 300000
#define E_TOT   (N_EDGES + N_NODES)
#define F_IN    79
#define HEADS   10
#define CH      112
#define HC      1120
#define D1      54
#define NG      64
#define BN_EPS  1e-5f

// ---- monotone float<->uint ordering for atomicMax on floats ----
__device__ __forceinline__ unsigned f2ord(float f) {
    int i = __float_as_int(f);
    return (i >= 0) ? ((unsigned)i | 0x80000000u) : ~(unsigned)i;
}
__device__ __forceinline__ float ord2f(unsigned u) {
    int i = (u & 0x80000000u) ? (int)(u ^ 0x80000000u) : (int)~u;
    return __int_as_float(i);
}

// ---------------- fills ----------------
__global__ void k_fill_f32(float* p, float v, int n) {
    int i = blockIdx.x * blockDim.x + threadIdx.x;
    if (i < n) p[i] = v;
}
__global__ void k_fill_u32(unsigned* p, unsigned v, int n) {
    int i = blockIdx.x * blockDim.x + threadIdx.x;
    if (i < n) p[i] = v;
}

// ---------------- WMMA GEMM: C[M,Nn] = A[M,K] @ W[K,Nn] ----------------
// 256 threads = 8 waves. Each wave owns a 32x32 output tile:
// 4 accumulators from 2 A-fragments x 2 B-fragments (4 WMMAs / K-step).
// Block tile: 256 rows x 32 cols. K staged into double-buffered LDS as f16
// (A row-major, B transposed [n][k]) so fragments are ds_load_b128.
// Staging: batched loads into distinct regs (clause + single wait), then
// convert+store. All global loads unconditional (clamped addr + cndmask).

template<int K>
__device__ __forceinline__ void stage_tile(const float* __restrict__ A,
                                           const float* __restrict__ W,
                                           int M, int Nn, int rowBase, int colBase,
                                           int kk, bool pf,
                                           _Float16* __restrict__ sA,
                                           _Float16* __restrict__ sB, int tid)
{
    // ---- A tile 256x32 = 4096 half2-pairs; 16 pairs/thread in 2 batches ----
    #pragma unroll
    for (int g = 0; g < 2; ++g) {
        float v[16];
        int   ldsOff[8];
        #pragma unroll
        for (int i = 0; i < 8; ++i) {
            int p  = tid + (g * 8 + i) * 256;   // pair index 0..4095
            int r  = p >> 4;                    // row 0..255
            int k2 = (p & 15) * 2;              // even k 0..30
            int gr = rowBase + r; gr = (gr < M) ? gr : (M - 1);
            int k0 = kk + k2;
            const float* ap = &A[gr * K];
            v[2 * i]     = ap[(k0 < K) ? k0 : (K - 1)];
            v[2 * i + 1] = ap[(k0 + 1 < K) ? (k0 + 1) : (K - 1)];
            ldsOff[i] = r * 32 + k2;
            if (pf && i == 0) __builtin_prefetch(&ap[k0 + 32], 0, 3); // speculative
        }
        #pragma unroll
        for (int i = 0; i < 8; ++i) {
            int p  = tid + (g * 8 + i) * 256;
            int k0 = kk + (p & 15) * 2;
            h2 pk;
            pk.x = (_Float16)((k0 < K)     ? v[2 * i]     : 0.0f);
            pk.y = (_Float16)((k0 + 1 < K) ? v[2 * i + 1] : 0.0f);
            *(h2*)&sA[ldsOff[i]] = pk;
        }
    }
    // ---- B tile transposed 32(n) x 32(k) = 512 pairs; 2 pairs/thread ----
    {
        float v[4];
        int   ldsOff[2];
        #pragma unroll
        for (int i = 0; i < 2; ++i) {
            int p  = tid + i * 256;             // 0..511
            int k2 = (p >> 5) * 2;              // even k 0..30
            int n  = p & 31;
            int k0 = kk + k2;
            v[2 * i]     = W[((k0 < K)     ? k0     : (K - 1)) * Nn + colBase + n];
            v[2 * i + 1] = W[((k0 + 1 < K) ? k0 + 1 : (K - 1)) * Nn + colBase + n];
            ldsOff[i] = n * 32 + k2;
        }
        #pragma unroll
        for (int i = 0; i < 2; ++i) {
            int p  = tid + i * 256;
            int k0 = kk + (p >> 5) * 2;
            h2 pk;
            pk.x = (_Float16)((k0 < K)     ? v[2 * i]     : 0.0f);
            pk.y = (_Float16)((k0 + 1 < K) ? v[2 * i + 1] : 0.0f);
            *(h2*)&sB[ldsOff[i]] = pk;
        }
    }
}

__device__ __forceinline__ v16h frag16(const _Float16* p0, const _Float16* p1)
{
    h8 lo = *(const h8*)p0;
    h8 hi = *(const h8*)p1;
    return __builtin_shufflevector(lo, hi, 0,1,2,3,4,5,6,7,8,9,10,11,12,13,14,15);
}

template<int K, int KP>
__global__ void __launch_bounds__(256)
k_wmma_gemm(const float* __restrict__ A, const float* __restrict__ W,
            float* __restrict__ C, int M, int Nn)
{
    __shared__ _Float16 sA[2][256 * 32];   // [row][k], double buffered
    __shared__ _Float16 sB[2][32 * 32];    // [n][k] transposed, double buffered

    const int tid  = threadIdx.x;
    const int wave = tid >> 5;             // 0..7 -> 32 rows each
    const int lane = tid & 31;
    const int rowBase = blockIdx.x * 256;
    const int colBase = blockIdx.y * 32;

    const int mrow = lane & 15;
    const int hsel = lane >> 4;            // half-wave select
    const int ncl  = lane & 15;            // fragment column within 16

    v8f acc00 = {}, acc01 = {}, acc10 = {}, acc11 = {};

    stage_tile<K>(A, W, M, Nn, rowBase, colBase, 0, (32 < KP), sA[0], sB[0], tid);
    __syncthreads();

    int pb = 0;
    for (int kk = 0; kk < KP; kk += 32) {
        if (kk + 32 < KP)
            stage_tile<K>(A, W, M, Nn, rowBase, colBase, kk + 32,
                          (kk + 64 < KP), sA[pb ^ 1], sB[pb ^ 1], tid);

        // ---- fragments (ISA 7.12.2), each ds_load_b128 x2 ----
        const _Float16* ar0 = &sA[pb][(wave * 32 + mrow) * 32];       // rows +0..15
        const _Float16* ar1 = ar0 + 16 * 32;                          // rows +16..31
        v16h a0 = frag16(ar0 + hsel * 8, ar0 + 16 + hsel * 8);
        v16h a1 = frag16(ar1 + hsel * 8, ar1 + 16 + hsel * 8);
        const _Float16* br0 = &sB[pb][ncl * 32 + hsel * 8];           // cols 0..15
        const _Float16* br1 = &sB[pb][(ncl + 16) * 32 + hsel * 8];    // cols 16..31
        v16h b0 = frag16(br0, br0 + 16);
        v16h b1 = frag16(br1, br1 + 16);

        acc00 = __builtin_amdgcn_wmma_f32_16x16x32_f16(false, a0, false, b0,
                                                       (short)0, acc00, false, false);
        acc01 = __builtin_amdgcn_wmma_f32_16x16x32_f16(false, a0, false, b1,
                                                       (short)0, acc01, false, false);
        acc10 = __builtin_amdgcn_wmma_f32_16x16x32_f16(false, a1, false, b0,
                                                       (short)0, acc10, false, false);
        acc11 = __builtin_amdgcn_wmma_f32_16x16x32_f16(false, a1, false, b1,
                                                       (short)0, acc11, false, false);
        __syncthreads();
        pb ^= 1;
    }

    // C/D layout: VGPR r, lanes 0-15 -> M=r ; lanes 16-31 -> M=8+r ; N=lane&15
    const int mb  = hsel * 8;
    const int gc0 = colBase + ncl;
    const int gc1 = gc0 + 16;
    #pragma unroll
    for (int r = 0; r < 8; ++r) {
        int gr0 = rowBase + wave * 32 + mb + r;      // rows 0..15 sub-tile
        int gr1 = gr0 + 16;                          // rows 16..31 sub-tile
        if (gr0 < M) { C[gr0 * Nn + gc0] = acc00[r]; C[gr0 * Nn + gc1] = acc01[r]; }
        if (gr1 < M) { C[gr1 * Nn + gc0] = acc10[r]; C[gr1 * Nn + gc1] = acc11[r]; }
    }
}

// ---------------- attention scores: one wave per (node, head) ----------------
__global__ void k_scores(const float* __restrict__ h, const float* __restrict__ asrc,
                         const float* __restrict__ adst,
                         float* __restrict__ s_src, float* __restrict__ s_dst)
{
    int gwave = (blockIdx.x * blockDim.x + threadIdx.x) >> 5;
    int lane  = threadIdx.x & 31;
    if (gwave >= N_NODES * HEADS) return;
    int node = gwave / HEADS, head = gwave % HEADS;
    const float* hp = h + node * HC + head * CH;
    const float* as = asrc + head * CH;
    const float* ad = adst + head * CH;
    float ss = 0.f, sd = 0.f;
    for (int c = lane; c < CH; c += 32) { float v = hp[c]; ss += v * as[c]; sd += v * ad[c]; }
    #pragma unroll
    for (int off = 16; off > 0; off >>= 1) {
        ss += __shfl_xor(ss, off, 32);
        sd += __shfl_xor(sd, off, 32);
    }
    if (lane == 0) { s_src[gwave] = ss; s_dst[gwave] = sd; }
}

// ---------------- edge pass 1: leaky-relu logit + segment max ----------------
__global__ void k_edge_logit_max(const int* __restrict__ ei,
                                 const float* __restrict__ s_src, const float* __restrict__ s_dst,
                                 float* __restrict__ e_p, unsigned* __restrict__ mmax)
{
    int i = blockIdx.x * blockDim.x + threadIdx.x;
    if (i >= E_TOT * HEADS) return;
    int e = i / HEADS, hd = i % HEADS;
    int s, d;
    if (e < N_EDGES) { s = ei[e]; d = ei[N_EDGES + e]; } else { s = d = e - N_EDGES; }
    float x = s_src[s * HEADS + hd] + s_dst[d * HEADS + hd];
    x = (x > 0.f) ? x : 0.2f * x;
    e_p[i] = x;
    atomicMax(&mmax[d * HEADS + hd], f2ord(x));
}

// ---------------- edge pass 2: exp(logit - max), segment sum ----------------
__global__ void k_edge_exp_sum(const int* __restrict__ ei, const unsigned* __restrict__ mmax,
                               float* __restrict__ e_p, float* __restrict__ denom)
{
    int i = blockIdx.x * blockDim.x + threadIdx.x;
    if (i >= E_TOT * HEADS) return;
    int e = i / HEADS, hd = i % HEADS;
    int d;
    if (e < N_EDGES) { d = ei[N_EDGES + e]; } else { d = e - N_EDGES; }
    float m = ord2f(mmax[d * HEADS + hd]);
    float p = expf(e_p[i] - m);
    e_p[i] = p;
    atomicAdd(&denom[d * HEADS + hd], p);
}

// ---------------- edge pass 3: out[dst] += alpha * h[src] ----------------
__global__ void __launch_bounds__(256)
k_edge_aggregate(const int* __restrict__ ei, const float* __restrict__ e_p,
                 const float* __restrict__ denom,
                 const float* __restrict__ h, float* __restrict__ out)
{
    int e = blockIdx.x;
    int s, d;
    if (e < N_EDGES) { s = ei[e]; d = ei[N_EDGES + e]; } else { s = d = e - N_EDGES; }
    const float* hs = h + s * HC;
    float*       od = out + d * HC;
    for (int c = threadIdx.x; c < HC; c += blockDim.x) {
        int hd = c / CH;
        float alpha = e_p[e * HEADS + hd] / (denom[d * HEADS + hd] + 1e-16f);
        atomicAdd(&od[c], alpha * hs[c]);
    }
}

// ---------------- fused bias + ReLU + BatchNorm (in place) ----------------
__global__ void k_post(float* __restrict__ buf, const float* __restrict__ bias,
                       const float* __restrict__ w, const float* __restrict__ b,
                       const float* __restrict__ mean, const float* __restrict__ var)
{
    int i = blockIdx.x * blockDim.x + threadIdx.x;
    if (i >= N_NODES * HC) return;
    int c = i % HC;
    float v = buf[i] + bias[c];
    v = (v > 0.f) ? v : 0.f;
    v = (v - mean[c]) * (w[c] * rsqrtf(var[c] + BN_EPS)) + b[c];
    buf[i] = v;
}

// ---------------- global mean pool (sums + counts) ----------------
__global__ void k_pool(const int* __restrict__ batch, const float* __restrict__ feat,
                       float* __restrict__ pooled, float* __restrict__ cnt)
{
    int n = blockIdx.x;
    int g = batch[n];
    const float* f = feat + n * HC;
    float*       p = pooled + g * HC;
    for (int c = threadIdx.x; c < HC; c += blockDim.x) atomicAdd(&p[c], f[c]);
    if (threadIdx.x == 0) atomicAdd(&cnt[g], 1.0f);
}

// ---------------- MLP head: one block per graph ----------------
__global__ void k_mlp(const float* __restrict__ pooled, const float* __restrict__ cnt,
                      const float* __restrict__ l1w, const float* __restrict__ l1b,
                      const float* __restrict__ l2w, const float* __restrict__ l2b,
                      float* __restrict__ out)
{
    __shared__ float mean[HC];
    __shared__ float z[D1];
    int g = blockIdx.x;
    float inv = 1.0f / fmaxf(cnt[g], 1.0f);
    for (int c = threadIdx.x; c < HC; c += blockDim.x) mean[c] = pooled[g * HC + c] * inv;
    __syncthreads();
    for (int j = threadIdx.x; j < D1; j += blockDim.x) {
        float acc = l1b[j];
        for (int c = 0; c < HC; ++c) acc += mean[c] * l1w[c * D1 + j];
        z[j] = (acc > 0.f) ? acc : 0.f;
    }
    __syncthreads();
    if (threadIdx.x == 0) {
        float acc = l2b[0];
        for (int j = 0; j < D1; ++j) acc += z[j] * l2w[j];
        out[g] = 1.0f / (1.0f + expf(-acc));
    }
}

extern "C" void kernel_launch(void* const* d_in, const int* in_sizes, int n_in,
                              void* d_out, int out_size, void* d_ws, size_t ws_size,
                              hipStream_t stream)
{
    const float* x        = (const float*)d_in[0];
    const int*   ei       = (const int*)  d_in[1];
    const int*   batch    = (const int*)  d_in[2];
    const float* W1       = (const float*)d_in[3];
    const float* att_src1 = (const float*)d_in[4];
    const float* att_dst1 = (const float*)d_in[5];
    const float* b1       = (const float*)d_in[6];
    const float* bn1_w    = (const float*)d_in[7];
    const float* bn1_b    = (const float*)d_in[8];
    const float* bn1_mean = (const float*)d_in[9];
    const float* bn1_var  = (const float*)d_in[10];
    const float* W2       = (const float*)d_in[11];
    const float* att_src2 = (const float*)d_in[12];
    const float* att_dst2 = (const float*)d_in[13];
    const float* b2       = (const float*)d_in[14];
    const float* bn2_w    = (const float*)d_in[15];
    const float* bn2_b    = (const float*)d_in[16];
    const float* bn2_mean = (const float*)d_in[17];
    const float* bn2_var  = (const float*)d_in[18];
    const float* lin1_w   = (const float*)d_in[19];
    const float* lin1_b   = (const float*)d_in[20];
    const float* lin2_w   = (const float*)d_in[21];
    const float* lin2_b   = (const float*)d_in[22];
    float* out = (float*)d_out;

    // workspace carve-out
    char* ws = (char*)d_ws;
    size_t off = 0;
    auto carve = [&](size_t bytes) -> void* {
        void* p = ws + off;
        off += (bytes + 255) & ~(size_t)255;
        return p;
    };
    float*    bufA   = (float*)   carve((size_t)N_NODES * HC * sizeof(float)); // GEMM outputs h
    float*    bufB   = (float*)   carve((size_t)N_NODES * HC * sizeof(float)); // aggregation / features
    float*    e_p    = (float*)   carve((size_t)E_TOT * HEADS * sizeof(float));
    float*    s_src  = (float*)   carve((size_t)N_NODES * HEADS * sizeof(float));
    float*    s_dst  = (float*)   carve((size_t)N_NODES * HEADS * sizeof(float));
    unsigned* mmax   = (unsigned*)carve((size_t)N_NODES * HEADS * sizeof(unsigned));
    float*    denom  = (float*)   carve((size_t)N_NODES * HEADS * sizeof(float));
    float*    pooled = (float*)   carve((size_t)NG * HC * sizeof(float));
    float*    cnt    = (float*)   carve((size_t)NG * sizeof(float));

    const int TPB   = 256;
    const int nNH   = N_NODES * HEADS;                   // 300000
    const int nEH   = E_TOT * HEADS;                     // 3300000
    const int nFeat = N_NODES * HC;                      // 33600000
    dim3 gemmGrid((N_NODES + 255) / 256, HC / 32);       // (118, 35)
    int scoreBlocks = (nNH * 32 + TPB - 1) / TPB;

    // ================= Layer 1 =================
    k_wmma_gemm<F_IN, 96><<<gemmGrid, TPB, 0, stream>>>(x, W1, bufA, N_NODES, HC);
    k_scores<<<scoreBlocks, TPB, 0, stream>>>(bufA, att_src1, att_dst1, s_src, s_dst);
    k_fill_u32<<<(nNH + TPB - 1) / TPB, TPB, 0, stream>>>(mmax, 0u, nNH);
    k_fill_f32<<<(nNH + TPB - 1) / TPB, TPB, 0, stream>>>(denom, 0.f, nNH);
    k_fill_f32<<<(nFeat + TPB - 1) / TPB, TPB, 0, stream>>>(bufB, 0.f, nFeat);
    k_edge_logit_max<<<(nEH + TPB - 1) / TPB, TPB, 0, stream>>>(ei, s_src, s_dst, e_p, mmax);
    k_edge_exp_sum<<<(nEH + TPB - 1) / TPB, TPB, 0, stream>>>(ei, mmax, e_p, denom);
    k_edge_aggregate<<<E_TOT, TPB, 0, stream>>>(ei, e_p, denom, bufA, bufB);
    k_post<<<(nFeat + TPB - 1) / TPB, TPB, 0, stream>>>(bufB, b1, bn1_w, bn1_b, bn1_mean, bn1_var);

    // ================= Layer 2 =================
    k_wmma_gemm<HC, HC><<<gemmGrid, TPB, 0, stream>>>(bufB, W2, bufA, N_NODES, HC);
    k_scores<<<scoreBlocks, TPB, 0, stream>>>(bufA, att_src2, att_dst2, s_src, s_dst);
    k_fill_u32<<<(nNH + TPB - 1) / TPB, TPB, 0, stream>>>(mmax, 0u, nNH);
    k_fill_f32<<<(nNH + TPB - 1) / TPB, TPB, 0, stream>>>(denom, 0.f, nNH);
    k_fill_f32<<<(nFeat + TPB - 1) / TPB, TPB, 0, stream>>>(bufB, 0.f, nFeat); // bufB free after GEMM2
    k_edge_logit_max<<<(nEH + TPB - 1) / TPB, TPB, 0, stream>>>(ei, s_src, s_dst, e_p, mmax);
    k_edge_exp_sum<<<(nEH + TPB - 1) / TPB, TPB, 0, stream>>>(ei, mmax, e_p, denom);
    k_edge_aggregate<<<E_TOT, TPB, 0, stream>>>(ei, e_p, denom, bufA, bufB);
    k_post<<<(nFeat + TPB - 1) / TPB, TPB, 0, stream>>>(bufB, b2, bn2_w, bn2_b, bn2_mean, bn2_var);

    // ================= Pool + MLP head =================
    k_fill_f32<<<(NG * HC + TPB - 1) / TPB, TPB, 0, stream>>>(pooled, 0.f, NG * HC);
    k_fill_f32<<<1, 64, 0, stream>>>(cnt, 0.f, NG);
    k_pool<<<N_NODES, TPB, 0, stream>>>(batch, bufB, pooled, cnt);
    k_mlp<<<NG, 128, 0, stream>>>(pooled, cnt, lin1_w, lin1_b, lin2_w, lin2_b, out);
}
